// CausalSelfAttention_5557687681798
// MI455X (gfx1250) — matrix operand order repaired and checked by reading, not compiled, not run on hardware
//
#include <hip/hip_runtime.h>
#include <stdint.h>

// ---------------------------------------------------------------------------
// CDNA5 (gfx1250) bf16 WMMA attention forward.
//   B=2, T=2048, C=1024, H=16, D=64, tokens N = B*T = 4096.
// All matmuls are D = A * B^T with A,B row-major, K contiguous ->
// both WMMA operands load as contiguous 16B chunks per lane.
// 1/sqrt(D) = 0.125 is baked exactly (power of two) into the Q projection.
// Softmax row-sums computed with a WMMA against an all-ones B fragment.
// Flash loop: K fragments software-pipelined one iteration ahead.
// ---------------------------------------------------------------------------

typedef __attribute__((ext_vector_type(16))) __bf16        v16bf;
typedef __attribute__((ext_vector_type(8)))  float         v8f;
typedef __attribute__((ext_vector_type(4)))  unsigned int  v4u;

union Frag { v16bf v; v4u u[2]; };
union VecF { v8f  v; float f[8]; };

__device__ __forceinline__ unsigned short f32_to_bf16u(float x) {
  union { float f; unsigned int u; } c; c.f = x;
  unsigned int u = c.u;
  unsigned int r = u + 0x7FFFu + ((u >> 16) & 1u);   // round-to-nearest-even
  return (unsigned short)(r >> 16);
}

// A-operand (16x32, M x K): lane<16 -> K {k0..k0+7, k0+16..k0+23}
//                           lane>=16 -> K {k0+8..+15, k0+24..+31}
__device__ __forceinline__ Frag load_a(const unsigned short* base, int ld,
                                       int row0, int k0, int lane) {
  int r = row0 + (lane & 15);
  int k = k0 + ((lane >> 4) << 3);            // +0 or +8 elements
  const v4u* p = (const v4u*)(base + (size_t)r * ld + k);
  Frag f; f.u[0] = p[0]; f.u[1] = p[2];       // +0 and +16 elements
  return f;
}

// B-operand (32x16, K x N), sourced from row-major [N, K] matrix (B = src^T):
// lane<16 -> K {k0..k0+7, k0+16..k0+23}; lane>=16 -> K {k0+8..+15, k0+24..+31}
__device__ __forceinline__ Frag load_b(const unsigned short* base, int ld,
                                       int n0, int k0, int lane) {
  int r = n0 + (lane & 15);
  int k = k0 + ((lane >> 4) << 3);            // +0 or +8 elements
  const v4u* p = (const v4u*)(base + (size_t)r * ld + k);
  Frag f; f.u[0] = p[0]; f.u[1] = p[2];
  return f;
}

__device__ __forceinline__ v8f wmma_bf16(const Frag& a, const Frag& b, v8f c) {
  // 8 args: (neg_a, A, neg_b, B, c_mod, C, reuse_a, reuse_b)
  return __builtin_amdgcn_wmma_f32_16x16x32_bf16(
      false, a.v, false, b.v, (short)0, c, false, false);
}

// XOR-lane max via ds_swizzle (group-of-32 mode: and=0x1F, or=0, xor=XM).
// Masks 1/2/4/8 stay within each 16-lane half (matches C-layout row split).
template <int XM>
__device__ __forceinline__ float swz_max(float v) {
  int s = __builtin_amdgcn_ds_swizzle(__float_as_int(v), (XM << 10) | 0x1F);
  return fmaxf(v, __int_as_float(s));
}

// ---------------------------------------------------------------------------
// f32 -> bf16 conversion
// ---------------------------------------------------------------------------
__global__ void cvt_bf16_kernel(const float* __restrict__ src,
                                unsigned short* __restrict__ dst, int n) {
  int i = blockIdx.x * blockDim.x + threadIdx.x;
  if (i < n) dst[i] = f32_to_bf16u(src[i]);
}

// ---------------------------------------------------------------------------
// GEMM: out = A[4096,1024] * W[1024,1024]^T   (both bf16, K contiguous)
// MODE 0: store bf16 as K head layout [B,H,T,D]
// MODE 1: store bf16 as Q head layout [B,H,T,D], scaled by 0.125 (exact)
// MODE 2: store bf16 as V transposed  [B,H,D,T]   (packed b128 stores)
// MODE 3: store f32 + bias to [4096,1024] (final projection)
// Workgroup = 8 waves, WG tile 128(M) x 64(N), wave tile 32x32.
// k-loop double-buffered (branch-free wrap-around preload).
// ---------------------------------------------------------------------------
template <int MODE>
__global__ __launch_bounds__(256) void gemm_bf16_kernel(
    const unsigned short* __restrict__ A, const unsigned short* __restrict__ W,
    unsigned short* __restrict__ obf, float* __restrict__ of,
    const float* __restrict__ bias) {
  const int lane = threadIdx.x & 31;
  const int wave = threadIdx.x >> 5;
  const int m0 = blockIdx.x * 128 + (wave >> 1) * 32;
  const int n0 = blockIdx.y * 64  + (wave & 1) * 32;

  v8f acc00 = {}, acc01 = {}, acc10 = {}, acc11 = {};

  Frag a0 = load_a(A, 1024, m0,      0, lane);
  Frag a1 = load_a(A, 1024, m0 + 16, 0, lane);
  Frag b0 = load_b(W, 1024, n0,      0, lane);
  Frag b1 = load_b(W, 1024, n0 + 16, 0, lane);

  for (int k = 0; k < 1024; k += 32) {
    const int kn = (k + 32) & 1023;            // wrap: last preload is harmless
    Frag na0 = load_a(A, 1024, m0,      kn, lane);
    Frag na1 = load_a(A, 1024, m0 + 16, kn, lane);
    Frag nb0 = load_b(W, 1024, n0,      kn, lane);
    Frag nb1 = load_b(W, 1024, n0 + 16, kn, lane);
    acc00 = wmma_bf16(a0, b0, acc00);
    acc01 = wmma_bf16(a0, b1, acc01);
    acc10 = wmma_bf16(a1, b0, acc10);
    acc11 = wmma_bf16(a1, b1, acc11);
    a0 = na0; a1 = na1; b0 = nb0; b1 = nb1;
  }

  const int col = lane & 15, half = lane >> 4;
  VecF cs[2][2];
  cs[0][0].v = acc00; cs[0][1].v = acc01; cs[1][0].v = acc10; cs[1][1].v = acc11;

  for (int i = 0; i < 2; ++i)
    for (int j = 0; j < 2; ++j) {
      if (MODE == 2) {
        // Transposed V: per lane, rows r=0..7 are consecutive t -> pack b128
        int mbase = m0 + i * 16 + 8 * half;    // multiple of 8 -> 16B aligned
        int n = n0 + j * 16 + col;
        int h = n >> 6, dd = n & 63;
        int b = mbase >> 11, t = mbase & 2047;
        v4u pk;
        for (int r = 0; r < 4; ++r) {
          unsigned int lo = f32_to_bf16u(cs[i][j].f[2 * r]);
          unsigned int hi = f32_to_bf16u(cs[i][j].f[2 * r + 1]);
          pk[r] = lo | (hi << 16);
        }
        *(v4u*)(obf + (((size_t)(b * 16 + h) * 64 + dd) * 2048 + t)) = pk;
      } else {
        for (int r = 0; r < 8; ++r) {
          int m = m0 + i * 16 + r + 8 * half;  // token index (C-layout row)
          int n = n0 + j * 16 + col;           // output channel
          float val = cs[i][j].f[r];
          if (MODE == 3) {
            of[(size_t)m * 1024 + n] = val + bias[n];
          } else {
            if (MODE == 1) val *= 0.125f;      // 1/sqrt(D), exact in bf16
            int h = n >> 6, dd = n & 63;       // channel -> (head, dim)
            int b = m >> 11, t = m & 2047;     // token   -> (batch, time)
            obf[((size_t)(b * 16 + h) * 2048 + t) * 64 + dd] = f32_to_bf16u(val);
          }
        }
      }
    }
}

// ---------------------------------------------------------------------------
// Flash attention (no causal mask, matching reference).
// One wave owns a 16-row query tile; streams keys in steps of 32.
// K fragments are prefetched ONE ITERATION ahead (loads issued right after
// the S WMMAs, consumed at the top of the next iteration -> loadcnt waits
// carry allowances instead of 0). V fragments load at the top of the
// iteration and are consumed at the bottom, covered by the softmax block.
// S = q k^T (4 wmma; Q pre-scaled), online max via ds_swizzle butterflies,
// P -> LDS (C->A transpose), row sums via P x ones (1 wmma),
// then O = O*alpha + P V (4 wmma, V stored transposed).
// ---------------------------------------------------------------------------
__global__ __launch_bounds__(256) void flash_attn_kernel(
    const unsigned short* __restrict__ Q, const unsigned short* __restrict__ K,
    const unsigned short* __restrict__ VT, unsigned short* __restrict__ Ybf) {
  __shared__ unsigned short plds[8 * 16 * 32];   // 8 waves x (16x32 bf16) = 8 KB
  const int lane = threadIdx.x & 31;
  const int wave = threadIdx.x >> 5;
  const int half = lane >> 4;
  const int col  = lane & 15;
  unsigned short* pbuf = &plds[wave * 16 * 32];

  const int task  = blockIdx.x * 8 + wave;       // 0..4095 = 32 heads * 128 qtiles
  const int ttile = task & 127;
  const int bh    = task >> 7;                   // b*16 + h
  const int t0    = ttile * 16;

  const unsigned short* q  = Q  + (size_t)bh * 2048 * 64;
  const unsigned short* kk = K  + (size_t)bh * 2048 * 64;
  const unsigned short* vt = VT + (size_t)bh * 64 * 2048;

  // Query fragments, resident for the whole loop (K-dim = D = 64 -> 2 frags)
  Frag qf0 = load_a(q, 64, t0, 0,  lane);
  Frag qf1 = load_a(q, 64, t0, 32, lane);

  // All-ones bf16 B fragment: rowsum(P) = P x ones lands replicated in C-layout
  Frag ones;
  {
    v4u o = {0x3F803F80u, 0x3F803F80u, 0x3F803F80u, 0x3F803F80u};
    ones.u[0] = o; ones.u[1] = o;
  }

  VecF O[4];                                     // 16 x 64 f32 accumulator
  for (int t = 0; t < 4; ++t) O[t].v = (v8f){};
  float Mrun[8], Lrun[8];
  for (int r = 0; r < 8; ++r) { Mrun[r] = -1e30f; Lrun[r] = 0.0f; }

  // Prime the K pipeline (iteration 0 fragments)
  Frag kf00 = load_b(kk, 64,  0, 0,  lane);
  Frag kf01 = load_b(kk, 64,  0, 32, lane);
  Frag kf10 = load_b(kk, 64, 16, 0,  lane);
  Frag kf11 = load_b(kk, 64, 16, 32, lane);

  for (int s0 = 0; s0 < 2048; s0 += 32) {
    // V for this step: consumed at the bottom, behind the softmax block.
    Frag vf0 = load_b(vt, 2048,  0, s0, lane);
    Frag vf1 = load_b(vt, 2048, 16, s0, lane);
    Frag vf2 = load_b(vt, 2048, 32, s0, lane);
    Frag vf3 = load_b(vt, 2048, 48, s0, lane);

    v8f S0 = {}, S1 = {};                        // K loaded last iteration
    S0 = wmma_bf16(qf0, kf00, S0);
    S0 = wmma_bf16(qf1, kf01, S0);
    S1 = wmma_bf16(qf0, kf10, S1);
    S1 = wmma_bf16(qf1, kf11, S1);

    // Prefetch next step's K fragments (consumed next iteration; wrap is a
    // harmless redundant load on the final step).
    const int sn = (s0 + 32) & 2047;
    Frag nk00 = load_b(kk, 64, sn,      0,  lane);
    Frag nk01 = load_b(kk, 64, sn,      32, lane);
    Frag nk10 = load_b(kk, 64, sn + 16, 0,  lane);
    Frag nk11 = load_b(kk, 64, sn + 16, 32, lane);

    VecF sc0, sc1; sc0.v = S0; sc1.v = S1;
    float alpha[8];
    for (int r = 0; r < 8; ++r) {                // row m = r + 8*half
      float a0 = sc0.f[r];                       // already scaled via Q proj
      float a1 = sc1.f[r];
      float tmax = fmaxf(a0, a1);                // reduce over 16 cols/half-wave
      tmax = swz_max<1>(tmax);
      tmax = swz_max<2>(tmax);
      tmax = swz_max<4>(tmax);
      tmax = swz_max<8>(tmax);
      float mnew = fmaxf(Mrun[r], tmax);
      alpha[r] = __expf(Mrun[r] - mnew);
      float p0 = __expf(a0 - mnew);
      float p1 = __expf(a1 - mnew);
      Mrun[r] = mnew;
      int m = r + 8 * half;                      // C-layout -> LDS row-major
      pbuf[m * 32 + col]      = f32_to_bf16u(p0);
      pbuf[m * 32 + 16 + col] = f32_to_bf16u(p1);
    }

    for (int t = 0; t < 4; ++t)                  // rescale running accumulator
      for (int r = 0; r < 8; ++r) O[t].f[r] *= alpha[r];

    asm volatile("s_wait_dscnt 0" ::: "memory"); // LDS stores visible to reads

    // Re-read P in A-operand layout (lane = query row, K = key column)
    Frag pf;
    {
      const v4u* pl = (const v4u*)(pbuf + (lane & 15) * 32);
      pf.u[0] = pl[half];                        // K 0..7  (or 8..15)
      pf.u[1] = pl[2 + half];                    // K 16..23 (or 24..31)
    }

    // Row sums of bf16 P via WMMA against ones (replaces 32 ds_bpermute)
    VecF psum; psum.v = wmma_bf16(pf, ones, (v8f){});
    for (int r = 0; r < 8; ++r)
      Lrun[r] = Lrun[r] * alpha[r] + psum.f[r];

    O[0].v = wmma_bf16(pf, vf0, O[0].v);         // O += P * V (V^T rows = d)
    O[1].v = wmma_bf16(pf, vf1, O[1].v);
    O[2].v = wmma_bf16(pf, vf2, O[2].v);
    O[3].v = wmma_bf16(pf, vf3, O[3].v);

    kf00 = nk00; kf01 = nk01; kf10 = nk10; kf11 = nk11;
  }

  // Epilogue: y = O / L, merge heads back into [token, C] bf16
  const int b = bh >> 4, h = bh & 15;
  for (int t = 0; t < 4; ++t)
    for (int r = 0; r < 8; ++r) {
      int m  = t0 + r + 8 * half;
      int ch = h * 64 + t * 16 + col;
      float val = O[t].f[r] / Lrun[r];
      Ybf[(size_t)(b * 2048 + m) * 1024 + ch] = f32_to_bf16u(val);
    }
}

// ---------------------------------------------------------------------------
// Launch: x,Wk,Wq,Wv,Wp -> bf16; 3 proj GEMMs; flash attention; out GEMM+bias.
// Workspace: 48 MB of bf16 staging.
// ---------------------------------------------------------------------------
extern "C" void kernel_launch(void* const* d_in, const int* in_sizes, int n_in,
                              void* d_out, int out_size, void* d_ws, size_t ws_size,
                              hipStream_t stream) {
  (void)in_sizes; (void)n_in; (void)out_size; (void)ws_size;

  const float* x  = (const float*)d_in[0];
  const float* Wk = (const float*)d_in[1];
  const float* Wq = (const float*)d_in[2];
  const float* Wv = (const float*)d_in[3];
  const float* Wp = (const float*)d_in[4];
  const float* bp = (const float*)d_in[5];
  float* out = (float*)d_out;

  char* ws = (char*)d_ws;
  const size_t NB  = (size_t)4096 * 1024 * 2;   // token-major bf16 tensor
  const size_t WB  = (size_t)1024 * 1024 * 2;   // weight bf16
  const size_t HB  = (size_t)2 * 16 * 2048 * 64 * 2; // head-major bf16 tensor
  unsigned short* xbf  = (unsigned short*)ws; ws += NB;
  unsigned short* wkbf = (unsigned short*)ws; ws += WB;
  unsigned short* wqbf = (unsigned short*)ws; ws += WB;
  unsigned short* wvbf = (unsigned short*)ws; ws += WB;
  unsigned short* wpbf = (unsigned short*)ws; ws += WB;
  unsigned short* Qbf  = (unsigned short*)ws; ws += HB;
  unsigned short* Kbf  = (unsigned short*)ws; ws += HB;
  unsigned short* VTbf = (unsigned short*)ws; ws += HB;
  unsigned short* Ybf  = (unsigned short*)ws; ws += NB;

  const int nX = 4096 * 1024, nW = 1024 * 1024;
  cvt_bf16_kernel<<<(nX + 255) / 256, 256, 0, stream>>>(x,  xbf,  nX);
  cvt_bf16_kernel<<<(nW + 255) / 256, 256, 0, stream>>>(Wk, wkbf, nW);
  cvt_bf16_kernel<<<(nW + 255) / 256, 256, 0, stream>>>(Wq, wqbf, nW);
  cvt_bf16_kernel<<<(nW + 255) / 256, 256, 0, stream>>>(Wv, wvbf, nW);
  cvt_bf16_kernel<<<(nW + 255) / 256, 256, 0, stream>>>(Wp, wpbf, nW);

  dim3 g(32, 16), blk(256);
  gemm_bf16_kernel<1><<<g, blk, 0, stream>>>(xbf, wqbf, Qbf,  nullptr, nullptr);
  gemm_bf16_kernel<0><<<g, blk, 0, stream>>>(xbf, wkbf, Kbf,  nullptr, nullptr);
  gemm_bf16_kernel<2><<<g, blk, 0, stream>>>(xbf, wvbf, VTbf, nullptr, nullptr);

  flash_attn_kernel<<<512, 256, 0, stream>>>(Qbf, Kbf, VTbf, Ybf);

  gemm_bf16_kernel<3><<<g, blk, 0, stream>>>(Ybf, wpbf, nullptr, out, bp);
}
